// PointConv_87050397155931
// MI455X (gfx1250) — matrix operand order repaired and checked by reading, not compiled, not run on hardware
//
#include <hip/hip_runtime.h>

// ---------------------------------------------------------------------------
// PointConv forward on MI455X (gfx1250, wave32, WMMA bf16).
//  inputs: x[N,64] f32, pos[N,3] f32, W1[67,128], b1[128], W2[128,128],
//          b2[128], Wg[128,256], bg[256], edge_index[2,E] int64
//  out:    relu( segment_max( edgeMLP ) @ Wg + bg )  -> [N,256] f32
// ---------------------------------------------------------------------------

typedef __attribute__((ext_vector_type(16))) __bf16 v16bf;
typedef __attribute__((ext_vector_type(8)))  float  v8f;

struct alignas(16) U4 { unsigned int x, y, z, w; };

union FragB16 { v16bf v; U4 q[2]; };

#define FDIM  64
#define K1PAD 96
#define HDIM  128
#define GDIM  256

__device__ __forceinline__ unsigned short f32_to_bf16_bits(float f) {
  unsigned int u = __float_as_uint(f);
  u += 0x7FFFu + ((u >> 16) & 1u);   // round to nearest even
  return (unsigned short)(u >> 16);
}

__device__ __forceinline__ void lds_fence() {
  asm volatile("s_wait_dscnt 0" ::: "memory");
}

__device__ __forceinline__ v8f vzero8() {
  v8f z = {0.f, 0.f, 0.f, 0.f, 0.f, 0.f, 0.f, 0.f};
  return z;
}

#define WMMA_BF16(a, b, c) \
  __builtin_amdgcn_wmma_f32_16x16x32_bf16(false, (a), false, (b), (short)0, (c), false, false)

// B fragment: prepacked layout [kt][nt][lane(32)][i(16)] bf16 -> 32B per lane.
__device__ __forceinline__ v16bf load_b_frag(const unsigned short* __restrict__ B,
                                             int kc, int nt, int NT, int lane) {
  FragB16 f;
  const U4* p = (const U4*)(B + (((size_t)(kc * NT + nt)) << 9) + (lane << 4));
  f.q[0] = p[0];
  f.q[1] = p[1];
  return f.v;
}

// A fragment from a bf16 row in LDS: lane holds row m=lane&15,
// K = kc*32 + base..base+7 and base+16..base+23, base = (lane>>4)*8.
__device__ __forceinline__ v16bf load_a_frag_lds(const unsigned short* row, int kc, int hi) {
  FragB16 f;
  const U4* p = (const U4*)(row + kc * 32 + hi * 8);
  f.q[0] = p[0];
  f.q[1] = p[2];   // +16 elements = +2 U4
  return f.v;
}

__device__ __forceinline__ U4 pack8_bf16(const float* p) {
  U4 q;
  q.x = (unsigned)f32_to_bf16_bits(p[0]) | ((unsigned)f32_to_bf16_bits(p[1]) << 16);
  q.y = (unsigned)f32_to_bf16_bits(p[2]) | ((unsigned)f32_to_bf16_bits(p[3]) << 16);
  q.z = (unsigned)f32_to_bf16_bits(p[4]) | ((unsigned)f32_to_bf16_bits(p[5]) << 16);
  q.w = (unsigned)f32_to_bf16_bits(p[6]) | ((unsigned)f32_to_bf16_bits(p[7]) << 16);
  return q;
}

// --------------------------- prep kernels ----------------------------------

__global__ void pc_prep_xh(const float* __restrict__ x, unsigned short* __restrict__ xh,
                           int total) {
  int i = blockIdx.x * blockDim.x + threadIdx.x;
  if (i < total) xh[i] = f32_to_bf16_bits(x[i]);
}

// Repack row-major W[K_in, NT*16] f32 into WMMA B-fragment bf16 layout
// dst[kt][nt][lane][i] = W[kt*32 + (i>>3)*16 + (lane>>4)*8 + (i&7)][nt*16 + (lane&15)]
__global__ void pc_prep_B(const float* __restrict__ W, int K_in, int KT, int NT,
                          unsigned short* __restrict__ dst) {
  int idx = blockIdx.x * blockDim.x + threadIdx.x;
  int total = KT * NT * 512;
  if (idx >= total) return;
  int i    = idx & 15;
  int lane = (idx >> 4) & 31;
  int nt   = (idx >> 9) % NT;
  int kt   = idx / (NT * 512);
  int k    = kt * 32 + (i >> 3) * 16 + (lane >> 4) * 8 + (i & 7);
  int nn   = nt * 16 + (lane & 15);
  float v  = (k < K_in) ? W[k * (NT * 16) + nn] : 0.0f;
  dst[idx] = f32_to_bf16_bits(v);
}

// --------------------------- edge MLP + scatter-max ------------------------

__global__ __launch_bounds__(256) void pc_edge_kernel(
    const unsigned short* __restrict__ xh,   // [N,64] bf16
    const float* __restrict__ pos,           // [N,3]
    const long long* __restrict__ rowp,      // target i
    const long long* __restrict__ colp,      // neighbor j
    const unsigned short* __restrict__ B1,   // 3x8 fragments
    const float* __restrict__ b1,
    const unsigned short* __restrict__ B2,   // 4x8 fragments
    const float* __restrict__ b2,
    unsigned int* __restrict__ agg,          // [N,128] f32-as-u32, init 0
    long long E, long long Et)
{
  __shared__ unsigned short ldsA[8][16 * K1PAD];   // per-wave 16x96 bf16
  __shared__ unsigned short ldsH[8][16 * HDIM];    // per-wave 16x128 bf16

  const int w    = threadIdx.x >> 5;
  const int lane = threadIdx.x & 31;
  const int m    = lane & 15;
  const int hi   = lane >> 4;
  const int n    = m;                 // C/D column owned by this lane

  long long tile = (long long)blockIdx.x * 8 + w;
  long long e0 = tile << 4;
  if (e0 >= Et) return;               // wave-uniform exit

  long long e = e0 + m;
  if (e >= Et) e = Et - 1;            // duplicate edge: harmless under max
  int r, c;
  if (e < E) { r = (int)rowp[e]; c = (int)colp[e]; }
  else       { r = c = (int)(e - E); }   // self loop

  // ---- stage edge-feature row [x[c] (64) | rel (3) | 0-pad] as bf16 ----
  unsigned short* rowA = &ldsA[w][m * K1PAD];
  const U4* xs = (const U4*)(xh + (size_t)c * FDIM);   // 8 x U4 per row
  if (hi == 0) {
    U4* dst = (U4*)rowA;                                // k = 0..47
    #pragma unroll
    for (int q = 0; q < 6; ++q) dst[q] = xs[q];
  } else {
    U4* dst = (U4*)(rowA + 48);                         // k = 48..63
    dst[0] = xs[6];
    dst[1] = xs[7];
    float r0 = pos[c * 3 + 0] - pos[r * 3 + 0];
    float r1 = pos[c * 3 + 1] - pos[r * 3 + 1];
    float r2 = pos[c * 3 + 2] - pos[r * 3 + 2];
    U4 pz, z;
    pz.x = (unsigned)f32_to_bf16_bits(r0) | ((unsigned)f32_to_bf16_bits(r1) << 16);
    pz.y = (unsigned)f32_to_bf16_bits(r2);
    pz.z = 0u; pz.w = 0u;
    z.x = z.y = z.z = z.w = 0u;
    U4* dst2 = (U4*)(rowA + 64);                        // k = 64..95
    dst2[0] = pz; dst2[1] = z; dst2[2] = z; dst2[3] = z;
  }
  lds_fence();

  // ---- layer 1: [16x96] @ [96x128] ----
  v8f acc[8];
  #pragma unroll
  for (int t = 0; t < 8; ++t) acc[t] = vzero8();

  const unsigned short* rA = &ldsA[w][m * K1PAD];
  #pragma unroll
  for (int kc = 0; kc < 3; ++kc) {
    v16bf a = load_a_frag_lds(rA, kc, hi);
    #pragma unroll
    for (int nt = 0; nt < 8; ++nt) {
      v16bf b = load_b_frag(B1, kc, nt, 8, lane);
      acc[nt] = WMMA_BF16(a, b, acc[nt]);
    }
  }

  // bias + relu, restage h1 as bf16 A-source
  unsigned short* hbase = &ldsH[w][0];
  #pragma unroll
  for (int nt = 0; nt < 8; ++nt) {
    float bv = b1[nt * 16 + n];
    #pragma unroll
    for (int r8 = 0; r8 < 8; ++r8) {
      float v = fmaxf(acc[nt][r8] + bv, 0.0f);
      int mr = r8 + hi * 8;
      hbase[mr * HDIM + nt * 16 + n] = f32_to_bf16_bits(v);
    }
  }
  lds_fence();

  // ---- layer 2: [16x128] @ [128x128] ----
  v8f acc2[8];
  #pragma unroll
  for (int t = 0; t < 8; ++t) acc2[t] = vzero8();

  const unsigned short* rH = &ldsH[w][m * HDIM];
  #pragma unroll
  for (int kc = 0; kc < 4; ++kc) {
    v16bf a = load_a_frag_lds(rH, kc, hi);
    #pragma unroll
    for (int nt = 0; nt < 8; ++nt) {
      v16bf b = load_b_frag(B2, kc, nt, 8, lane);
      acc2[nt] = WMMA_BF16(a, b, acc2[nt]);
    }
  }

  // bias + relu + scatter-max (post-relu values >= 0 -> u32 order == f32 order)
  #pragma unroll
  for (int nt = 0; nt < 8; ++nt) {
    float bv = b2[nt * 16 + n];
    #pragma unroll
    for (int r8 = 0; r8 < 8; ++r8) {
      float v = fmaxf(acc2[nt][r8] + bv, 0.0f);
      int mr = r8 + hi * 8;
      int node = __shfl(r, mr, 32);          // row index of edge mr in this tile
      atomicMax(agg + (size_t)node * HDIM + nt * 16 + n, __float_as_uint(v));
    }
  }
}

// --------------------------- global MLP ------------------------------------

__global__ __launch_bounds__(256) void pc_global_kernel(
    const float* __restrict__ agg,           // [N,128] (bit-identical u32 maxes)
    const unsigned short* __restrict__ Bg,   // 4x16 fragments
    const float* __restrict__ bg,
    float* __restrict__ out,                 // [N,256]
    int N)
{
  const int w    = threadIdx.x >> 5;
  const int lane = threadIdx.x & 31;
  const int m    = lane & 15;
  const int hi   = lane >> 4;
  const int n    = m;

  long long tile = (long long)blockIdx.x * 8 + w;
  int row0 = (int)(tile << 4);
  if (row0 >= N) return;                     // wave-uniform exit
  int myrow = row0 + m;
  if (myrow >= N) myrow = N - 1;             // clamped rows are not stored

  // keep all 4 A fragments of this 16-row tile resident
  v16bf a[4];
  #pragma unroll
  for (int kc = 0; kc < 4; ++kc) {
    const float* p = agg + (size_t)myrow * HDIM + kc * 32 + hi * 8;
    FragB16 f;
    f.q[0] = pack8_bf16(p);
    f.q[1] = pack8_bf16(p + 16);
    a[kc] = f.v;
  }

  #pragma unroll
  for (int nt = 0; nt < 16; ++nt) {
    v8f acc = vzero8();
    #pragma unroll
    for (int kc = 0; kc < 4; ++kc) {
      v16bf b = load_b_frag(Bg, kc, nt, 16, lane);
      acc = WMMA_BF16(a[kc], b, acc);
    }
    float bv = bg[nt * 16 + n];
    #pragma unroll
    for (int r8 = 0; r8 < 8; ++r8) {
      int mr = row0 + r8 + hi * 8;
      if (mr < N)
        out[(size_t)mr * GDIM + nt * 16 + n] = fmaxf(acc[r8] + bv, 0.0f);
    }
  }
}

// --------------------------- host launch -----------------------------------

extern "C" void kernel_launch(void* const* d_in, const int* in_sizes, int n_in,
                              void* d_out, int out_size, void* d_ws, size_t ws_size,
                              hipStream_t stream) {
  const float*     x   = (const float*)d_in[0];
  const float*     pos = (const float*)d_in[1];
  const float*     W1  = (const float*)d_in[2];
  const float*     b1  = (const float*)d_in[3];
  const float*     W2  = (const float*)d_in[4];
  const float*     b2  = (const float*)d_in[5];
  const float*     Wg  = (const float*)d_in[6];
  const float*     bg  = (const float*)d_in[7];
  const long long* ei  = (const long long*)d_in[8];
  float* out = (float*)d_out;

  const int       N  = in_sizes[0] / FDIM;
  const long long E  = (long long)in_sizes[8] / 2;
  const long long Et = E + N;

  // workspace carve-out (256B aligned slices)
  char* ws = (char*)d_ws;
  size_t off = 0;
  auto take = [&](size_t bytes) -> char* {
    char* p = ws + off;
    off = (off + bytes + 255) & ~(size_t)255;
    return p;
  };
  unsigned short* xh  = (unsigned short*)take((size_t)N * FDIM * 2);
  unsigned short* B1p = (unsigned short*)take((size_t)3 * 8  * 512 * 2);
  unsigned short* B2p = (unsigned short*)take((size_t)4 * 8  * 512 * 2);
  unsigned short* Bgp = (unsigned short*)take((size_t)4 * 16 * 512 * 2);
  unsigned int*   agg = (unsigned int*)take((size_t)N * HDIM * 4);

  // 1) precision/layout prep
  int totX = N * FDIM;
  pc_prep_xh<<<(totX + 255) / 256, 256, 0, stream>>>(x, xh, totX);
  pc_prep_B<<<(3 * 8  * 512 + 255) / 256, 256, 0, stream>>>(W1, FDIM + 3, 3, 8,  B1p);
  pc_prep_B<<<(4 * 8  * 512 + 255) / 256, 256, 0, stream>>>(W2, HDIM,     4, 8,  B2p);
  pc_prep_B<<<(4 * 16 * 512 + 255) / 256, 256, 0, stream>>>(Wg, HDIM,     4, 16, Bgp);

  // 2) agg init to 0 (valid identity: all contributions are post-ReLU >= 0)
  hipMemsetAsync(agg, 0, (size_t)N * HDIM * 4, stream);

  // 3) per-edge MLP + scatter-max  (16 edges per wave, 8 waves per block)
  long long tiles = (Et + 15) >> 4;
  int eblocks = (int)((tiles + 7) >> 3);
  pc_edge_kernel<<<eblocks, 256, 0, stream>>>(xh, pos, ei, ei + E,
                                              B1p, b1, B2p, b2, agg, E, Et);

  // 4) global MLP
  int ntiles = (N + 15) >> 4;
  pc_global_kernel<<<(ntiles + 7) >> 3, 256, 0, stream>>>((const float*)agg,
                                                          Bgp, bg, out, N);
}
